// BiasAttention_69698729279947
// MI455X (gfx1250) — matrix operand order repaired and checked by reading, not compiled, not run on hardware
//
#include <hip/hip_runtime.h>

// ---------------------------------------------------------------------------
// BiasAttention for MI455X (gfx1250): bf16 WMMA flash-attention pipeline.
//   B=8, N=2048, C=256, H=8, D=32
// Stages:
//   0) convert x / w_qkv / w_proj -> bf16 ws; sigmoid(atten_bias) -> fp32 ws
//   1) QKV GEMM (bf16 wmma, fp32 acc, 16x64 tile/wave) -> Q,K [B,H,N,D], Vt [B,H,D,N]
//   2) flash attention per (b,h,16 q-rows): v_wmma scores, online softmax,
//      P staged via LDS to A-fragment layout, v_wmma P*V -> O[B,N,C] bf16
//   3) out-proj GEMM (bf16 wmma, 16x64 tile/wave) + bias -> fp32 d_out
// ---------------------------------------------------------------------------

typedef __bf16 bf16_t;
typedef __attribute__((ext_vector_type(16))) __bf16 v16bf;
typedef __attribute__((ext_vector_type(8)))  __bf16 v8bf;
typedef __attribute__((ext_vector_type(8)))  float  v8f;

#define BB 8
#define NN 2048
#define CC 256
#define HH 8
#define DD 32
#define C3 768
#define MM (BB * NN)   // 16384

// workspace offsets (bytes)
#define OFF_XBF  ((size_t)0)          // 16384*256 bf16
#define OFF_WQBF ((size_t)8388608)    // 768*256 bf16
#define OFF_WPBF ((size_t)8781824)    // 256*256 bf16
#define OFF_Q    ((size_t)8912896)    // [B*H,2048,32] bf16
#define OFF_K    ((size_t)17301504)   // [B*H,2048,32] bf16
#define OFF_VT   ((size_t)25690112)   // [B*H,32,2048] bf16
#define OFF_OBF  ((size_t)34078720)   // [B*N,256] bf16
#define OFF_SB   ((size_t)42467328)   // [2048,2048] fp32 sigmoid(bias)

__device__ __forceinline__ v8f wmma_bf16(v16bf a, v16bf b, v8f c) {
  return __builtin_amdgcn_wmma_f32_16x16x32_bf16(false, a, false, b, (short)0, c,
                                                 false, false);
}

// A-fragment (16-bit, 16x32): caller passes p = rowbase + half*16 (+ kstep)
// elements: a[i] = p[i] (K=kb+i), a[8+i] = p[8+i] (K=kb+8+i)
template <typename P>
__device__ __forceinline__ v16bf ld_frag_a(P p) {
  v8bf lo = *(const v8bf*)(p);
  v8bf hi = *(const v8bf*)(p + 8);
  v16bf r;
#pragma unroll
  for (int i = 0; i < 8; ++i) { r[i] = lo[i]; r[8 + i] = hi[i]; }
  return r;
}

// B-fragment (16-bit, 32x16): lane holds column n, K = half*16 + i contiguous.
// caller passes p = colbase + half*16 (+ kstep): 16 contiguous bf16.
template <typename P>
__device__ __forceinline__ v16bf ld_frag_b(P p) {
  v8bf lo = *(const v8bf*)(p);
  v8bf hi = *(const v8bf*)(p + 8);
  v16bf r;
#pragma unroll
  for (int i = 0; i < 8; ++i) { r[i] = lo[i]; r[8 + i] = hi[i]; }
  return r;
}

// ---------------- stage 0: convert + sigmoid ----------------
__global__ void cvt_kernel(const float* __restrict__ x, const float* __restrict__ wq,
                           const float* __restrict__ wp, const float* __restrict__ bias,
                           bf16_t* __restrict__ xbf, bf16_t* __restrict__ wqbf,
                           bf16_t* __restrict__ wpbf, float* __restrict__ sb) {
  const long NX = (long)MM * CC;        // 4194304
  const long NWQ = (long)C3 * CC;       // 196608
  const long NWP = (long)CC * CC;       // 65536
  const long NB = (long)NN * NN;        // 4194304
  long tid = (long)blockIdx.x * blockDim.x + threadIdx.x;
  long stride = (long)gridDim.x * blockDim.x;
  for (long i = tid; i < NX; i += stride) xbf[i] = (bf16_t)x[i];
  for (long i = tid; i < NWQ; i += stride) wqbf[i] = (bf16_t)wq[i];
  for (long i = tid; i < NWP; i += stride) wpbf[i] = (bf16_t)wp[i];
  for (long i = tid; i < NB; i += stride) {
    float v = bias[i];
    sb[i] = 1.0f / (1.0f + exp2f(v * -1.44269504f));
  }
}

// ---------------- stage 1: QKV projection ----------------
// grid (M/16=1024, 3C/64=12), 32 threads; wave computes 16x64 tile
// (one shared A fragment per k-step feeds 4 WMMAs -> 4x less A traffic)
__global__ __launch_bounds__(32) void qkv_gemm(const bf16_t* __restrict__ xbf,
                                               const bf16_t* __restrict__ wqbf,
                                               bf16_t* __restrict__ Q,
                                               bf16_t* __restrict__ K,
                                               bf16_t* __restrict__ Vt) {
  const int lane = threadIdx.x;
  const int half = lane >> 4, lr = lane & 15;
  const int m0 = blockIdx.x * 16;
  const int c3b = blockIdx.y * 64;
  const bf16_t* ap = xbf + (size_t)(m0 + lr) * CC + half * 16;
  const bf16_t* bp = wqbf + (size_t)(c3b + lr) * CC + half * 16;
  v8f acc[4] = {{}, {}, {}, {}};
#pragma unroll
  for (int kk = 0; kk < CC; kk += 32) {
    v16bf a = ld_frag_a(ap + kk);
#pragma unroll
    for (int t = 0; t < 4; ++t) {
      v16bf b = ld_frag_b(bp + (size_t)t * 16 * CC + kk);
      acc[t] = wmma_bf16(a, b, acc[t]);
    }
  }
  const int b_ = m0 >> 11;          // / N
  const int n0 = m0 & (NN - 1);
#pragma unroll
  for (int t = 0; t < 4; ++t) {
    const int c3 = c3b + t * 16 + lr;
    const int which = c3 >> 8;      // 0=q 1=k 2=v
    const int c = c3 & 255;
    const int h = c >> 5, d = c & 31;
    const size_t bh = (size_t)(b_ * HH + h);
#pragma unroll
    for (int r = 0; r < 8; ++r) {
      const int n = n0 + r + 8 * half;
      bf16_t v = (bf16_t)acc[t][r];
      if (which == 0)      Q[(bh * NN + n) * DD + d] = v;
      else if (which == 1) K[(bh * NN + n) * DD + d] = v;
      else                 Vt[(bh * DD + d) * NN + n] = v;
    }
  }
}

// ---------------- stage 2: flash attention ----------------
// grid (N/16=128, B*H=64), 32 threads; wave owns 16 query rows, streams keys 32/iter
__global__ __launch_bounds__(32) void attn_kernel(const bf16_t* __restrict__ Q,
                                                  const bf16_t* __restrict__ K,
                                                  const bf16_t* __restrict__ Vt,
                                                  const float* __restrict__ sb,
                                                  bf16_t* __restrict__ O) {
  __shared__ bf16_t pL[16 * 32];   // P tile staging (C/D layout -> A layout)
  const int lane = threadIdx.x;
  const int half = lane >> 4, lr = lane & 15;
  const int q0 = blockIdx.x * 16;
  const int bh = blockIdx.y;
  const int b_ = bh >> 3, h = bh & 7;
  const bf16_t* Qb = Q + ((size_t)bh * NN + q0) * DD;
  const bf16_t* Kb = K + (size_t)bh * NN * DD;
  const bf16_t* Vb = Vt + (size_t)bh * DD * NN;
  const float* sbb = sb + (size_t)q0 * NN;

  // Q A-fragment: K-dim = D = 32, loaded once, reused for all 2048 keys
  v16bf aq = ld_frag_a(Qb + (size_t)lr * DD + half * 16);

  const float scale = 0.17677669529663687f;  // D^-0.5
  const float L2E = 1.44269504088896f;
  float mrun[8], lrun[8];
  v8f o0 = {}, o1 = {};
#pragma unroll
  for (int r = 0; r < 8; ++r) { mrun[r] = -3.0e38f; lrun[r] = 0.0f; }

  for (int j = 0; j < NN; j += 32) {
    // score WMMAs: B[d][n] = K[j+n][d] (row-major K => contiguous b128 loads)
    v16bf bk0 = ld_frag_b(Kb + (size_t)(j + lr) * DD + half * 16);
    v16bf bk1 = ld_frag_b(Kb + (size_t)(j + 16 + lr) * DD + half * 16);
    v8f s0 = {}, s1 = {};
    s0 = wmma_bf16(aq, bk0, s0);
    s1 = wmma_bf16(aq, bk1, s1);

#pragma unroll
    for (int r = 0; r < 8; ++r) {
      const int row = r + 8 * half;
      s0[r] = s0[r] * scale * sbb[(size_t)row * NN + j + lr];
      s1[r] = s1[r] * scale * sbb[(size_t)row * NN + j + 16 + lr];
    }
    // online softmax; row reduction across the 16 lanes holding one row
#pragma unroll
    for (int r = 0; r < 8; ++r) {
      float mx = fmaxf(s0[r], s1[r]);
#pragma unroll
      for (int off = 1; off < 16; off <<= 1)
        mx = fmaxf(mx, __shfl_xor(mx, off, 32));
      const float mnew = fmaxf(mrun[r], mx);
      const float f = exp2f((mrun[r] - mnew) * L2E);
      const float p0 = exp2f((s0[r] - mnew) * L2E);
      const float p1 = exp2f((s1[r] - mnew) * L2E);
      float rs = p0 + p1;
#pragma unroll
      for (int off = 1; off < 16; off <<= 1)
        rs += __shfl_xor(rs, off, 32);
      lrun[r] = lrun[r] * f + rs;
      mrun[r] = mnew;
      o0[r] *= f;
      o1[r] *= f;
      const int row = r + 8 * half;
      pL[row * 32 + lr] = (bf16_t)p0;        // keys j..j+15
      pL[row * 32 + 16 + lr] = (bf16_t)p1;   // keys j+16..j+31
    }
    // P as A-fragment from LDS (same-wave DS ops stay in order)
    v16bf apf = ld_frag_a(&pL[lr * 32 + half * 16]);
    // V B-fragments: B[kk][d] = Vt[d][j+kk] (transposed V => contiguous)
    v16bf bv0 = ld_frag_b(Vb + (size_t)lr * NN + j + half * 16);
    v16bf bv1 = ld_frag_b(Vb + (size_t)(16 + lr) * NN + j + half * 16);
    o0 = wmma_bf16(apf, bv0, o0);
    o1 = wmma_bf16(apf, bv1, o1);
  }
#pragma unroll
  for (int r = 0; r < 8; ++r) {
    const int row = r + 8 * half;
    const float inv = 1.0f / lrun[r];
    const size_t base = ((size_t)b_ * NN + q0 + row) * CC + h * DD;
    O[base + lr] = (bf16_t)(o0[r] * inv);
    O[base + 16 + lr] = (bf16_t)(o1[r] * inv);
  }
}

// ---------------- stage 3: output projection ----------------
// grid (M/16=1024, C/64=4), 32 threads; 16x64 tile per wave (A-frag reuse)
__global__ __launch_bounds__(32) void proj_gemm(const bf16_t* __restrict__ Obf,
                                                const bf16_t* __restrict__ wpbf,
                                                const float* __restrict__ bproj,
                                                float* __restrict__ out) {
  const int lane = threadIdx.x;
  const int half = lane >> 4, lr = lane & 15;
  const int m0 = blockIdx.x * 16;
  const int c3b = blockIdx.y * 64;
  const bf16_t* ap = Obf + (size_t)(m0 + lr) * CC + half * 16;
  const bf16_t* bp = wpbf + (size_t)(c3b + lr) * CC + half * 16;
  v8f acc[4] = {{}, {}, {}, {}};
#pragma unroll
  for (int kk = 0; kk < CC; kk += 32) {
    v16bf a = ld_frag_a(ap + kk);
#pragma unroll
    for (int t = 0; t < 4; ++t) {
      v16bf b = ld_frag_b(bp + (size_t)t * 16 * CC + kk);
      acc[t] = wmma_bf16(a, b, acc[t]);
    }
  }
#pragma unroll
  for (int t = 0; t < 4; ++t) {
    const int c3 = c3b + t * 16 + lr;
    const float bias = bproj[c3];
#pragma unroll
    for (int r = 0; r < 8; ++r) {
      const int m = m0 + r + 8 * half;
      out[(size_t)m * CC + c3] = acc[t][r] + bias;
    }
  }
}

extern "C" void kernel_launch(void* const* d_in, const int* in_sizes, int n_in,
                              void* d_out, int out_size, void* d_ws, size_t ws_size,
                              hipStream_t stream) {
  const float* x = (const float*)d_in[0];
  const float* atten_bias = (const float*)d_in[1];
  const float* w_qkv = (const float*)d_in[2];
  const float* w_proj = (const float*)d_in[3];
  const float* b_proj = (const float*)d_in[4];
  float* out = (float*)d_out;

  char* ws = (char*)d_ws;
  bf16_t* xbf = (bf16_t*)(ws + OFF_XBF);
  bf16_t* wqbf = (bf16_t*)(ws + OFF_WQBF);
  bf16_t* wpbf = (bf16_t*)(ws + OFF_WPBF);
  bf16_t* Q = (bf16_t*)(ws + OFF_Q);
  bf16_t* K = (bf16_t*)(ws + OFF_K);
  bf16_t* Vt = (bf16_t*)(ws + OFF_VT);
  bf16_t* Obf = (bf16_t*)(ws + OFF_OBF);
  float* sb = (float*)(ws + OFF_SB);

  cvt_kernel<<<dim3(512), dim3(256), 0, stream>>>(x, w_qkv, w_proj, atten_bias,
                                                  xbf, wqbf, wpbf, sb);
  qkv_gemm<<<dim3(MM / 16, C3 / 64), dim3(32), 0, stream>>>(xbf, wqbf, Q, K, Vt);
  attn_kernel<<<dim3(NN / 16, BB * HH), dim3(32), 0, stream>>>(Q, K, Vt, sb, Obf);
  proj_gemm<<<dim3(MM / 16, CC / 64), dim3(32), 0, stream>>>(Obf, wpbf, b_proj, out);
}
